// LanguageAttention_42039139893701
// MI455X (gfx1250) — matrix operand order repaired and checked by reading, not compiled, not run on hardware
//
#include <hip/hip_runtime.h>
#include <hip/hip_bf16.h>

#define BATCH  2
#define TOK    2048
#define DMODEL 1024
#define NHEADS 16
#define DHEAD  64
#define INNER  1024
#define MTOT   (BATCH * TOK)   // 4096

typedef _Float16 v16h __attribute__((ext_vector_type(16)));
typedef _Float16 v8h  __attribute__((ext_vector_type(8)));
typedef _Float16 v4h  __attribute__((ext_vector_type(4)));
typedef float    v8f  __attribute__((ext_vector_type(8)));
typedef float    v4f  __attribute__((ext_vector_type(4)));

static __device__ __forceinline__ int lane_id() { return (int)(threadIdx.x & 31u); }

// ---------------------------------------------------------------------------
// WMMA fragment loaders (CDNA5 16x16x32 f16, wave32).
//
// A-matrix 16x32 (MxK), source row-major with leading dim ld:
//   lanes 0-15 : row = lane,    K = k0 + {0..7, 16..23}
//   lanes 16-31: row = lane-16, K = k0 + {8..15, 24..31}
static __device__ __forceinline__ v16h load_fragA(const _Float16* base, int ld, int k0) {
  int l = lane_id();
  const _Float16* p = base + (size_t)(l & 15) * ld + k0 + ((l >> 4) << 3);
  v8h lo = *(const v8h*)(p);
  v8h hi = *(const v8h*)(p + 16);
  v16h f;
#pragma unroll
  for (int i = 0; i < 8; ++i) { f[i] = lo[i]; f[i + 8] = hi[i]; }
  return f;
}

// B-matrix 32x16 (KxN), source stored "column rows": row n holds all K for
// output column n (leading dim ld):
//   lanes 0-15 : col = lane,    K = k0 + {0..7, 16..23}
//   lanes 16-31: col = lane-16, K = k0 + {8..15, 24..31}
static __device__ __forceinline__ v16h load_fragB(const _Float16* base, int ld, int k0) {
  int l = lane_id();
  const _Float16* p = base + (size_t)(l & 15) * ld + k0 + ((l >> 4) << 3);
  v8h lo = *(const v8h*)(p);
  v8h hi = *(const v8h*)(p + 16);
  v16h f;
#pragma unroll
  for (int i = 0; i < 8; ++i) { f[i] = lo[i]; f[i + 8] = hi[i]; }
  return f;
}

static __device__ __forceinline__ v8f wmma_f16(v16h a, v16h b, v8f c) {
  return __builtin_amdgcn_wmma_f32_16x16x32_f16(false, a, false, b, (short)0, c, false, false);
}

// ---------------------------------------------------------------------------
// fp32 -> f16 conversion (vectorized x4)
__global__ void cvt_f32_to_f16_kernel(const float* __restrict__ in,
                                      _Float16* __restrict__ out, int n4) {
  int i = blockIdx.x * blockDim.x + threadIdx.x;
  if (i >= n4) return;
  v4f v = ((const v4f*)in)[i];
  v4h r;
#pragma unroll
  for (int k = 0; k < 4; ++k) r[k] = (_Float16)v[k];
  ((v4h*)out)[i] = r;
}

// ---------------------------------------------------------------------------
// Generic WMMA GEMM: Y[m][n] = sum_k A[m][k] * W[n][k]
//   A: MxK row-major f16, W: NxK row-major f16 (natural B-fragment layout).
// Per workgroup (256 thr = 8 waves): 64(M) x 256(N); per wave: 32(M) x 64(N).
// MODE 0: f16 store out[m*N + n]         (Q / K projections)
// MODE 1: f16 store transposed per head: out[((b*H+h)*64+c)*TOK + t]  (V^T)
// MODE 2: f32 store out[m*N + n]         (final output projection)
template <int MODE>
__global__ void gemm_wmma_kernel(const _Float16* __restrict__ A,
                                 const _Float16* __restrict__ W,
                                 void* __restrict__ outp,
                                 int M, int N, int K) {
  const int w  = threadIdx.x >> 5;
  const int wm = w & 1;          // 2 M-wave rows
  const int wn = w >> 1;         // 4 N-wave cols
  const int m0 = blockIdx.y * 64 + wm * 32;
  const int n0 = blockIdx.x * 256 + wn * 64;

  v8f acc[2][4];
#pragma unroll
  for (int i = 0; i < 2; ++i)
#pragma unroll
    for (int j = 0; j < 4; ++j) acc[i][j] = (v8f){0, 0, 0, 0, 0, 0, 0, 0};

  const _Float16* a0p = A + (size_t)m0 * K;
  const _Float16* a1p = A + (size_t)(m0 + 16) * K;

  for (int k0 = 0; k0 < K; k0 += 32) {
    if (k0 + 64 < K) {
      __builtin_prefetch(a0p + k0 + 64, 0, 0);   // global_prefetch_b8
      __builtin_prefetch(a1p + k0 + 64, 0, 0);
    }
    v16h a0 = load_fragA(a0p, K, k0);
    v16h a1 = load_fragA(a1p, K, k0);
    v16h b[4];
#pragma unroll
    for (int j = 0; j < 4; ++j)
      b[j] = load_fragB(W + (size_t)(n0 + 16 * j) * K, K, k0);
#pragma unroll
    for (int j = 0; j < 4; ++j) {
      acc[0][j] = wmma_f16(a0, b[j], acc[0][j]);
      acc[1][j] = wmma_f16(a1, b[j], acc[1][j]);
    }
  }

  const int l = lane_id();
  const int rsel = (l >> 4) << 3;   // 0 or 8
  const int col  = l & 15;
#pragma unroll
  for (int i = 0; i < 2; ++i)
#pragma unroll
    for (int j = 0; j < 4; ++j)
#pragma unroll
      for (int v = 0; v < 8; ++v) {
        int m = m0 + 16 * i + v + rsel;
        int n = n0 + 16 * j + col;
        float val = acc[i][j][v];
        if (MODE == 0) {
          ((_Float16*)outp)[(size_t)m * N + n] = (_Float16)val;
        } else if (MODE == 1) {
          int bb = m / TOK, t = m % TOK;
          int h = n >> 6, c = n & 63;
          ((_Float16*)outp)[((size_t)(bb * NHEADS + h) * DHEAD + c) * TOK + t] = (_Float16)val;
        } else {
          ((float*)outp)[(size_t)m * N + n] = val;
        }
      }
}

// ---------------------------------------------------------------------------
// Fused attention with online softmax, row-owner statistics.
// Grid: (TOK/64, NHEADS, BATCH), block 128 (4 waves). Each wave owns 16 q-rows,
// streams K/V in 32-key blocks: 4 WMMAs for scores, 4 WMMAs for P*V.
// Q,K layout: [b*TOK][INNER] (e = h*64+c). Vt layout: [b][h][c][TOK].
// Masked key columns get score 1e-9 (faithful to reference), softmax over keys.
//
// Softmax bookkeeping: lane l owns row (l&15), half (l>>4) of the 16x32 S tile
// staged in LDS. Per-row max/sum state lives in 2 VGPRs per lane; the two
// halves combine with a single __shfl_xor(...,16) each for max and sum.
// Per-row alpha / final 1/sum are redistributed to C-layout lanes via a tiny
// LDS array (contiguous ds_load_b128), avoiding ds_bpermute chains.
#define SLD 36   // padded f32 row stride (kills 2-way bank conflicts)
#define PLD 40   // padded f16 row stride (80B rows, 16B-aligned chunks)

__global__ void attention_wmma_kernel(const _Float16* __restrict__ Q,
                                      const _Float16* __restrict__ Kh,
                                      const _Float16* __restrict__ Vt,
                                      const int* __restrict__ mask,
                                      _Float16* __restrict__ AO) {
  __shared__ float    Sl[4][16][SLD];  // staged S tile (f32)
  __shared__ _Float16 Pl[4][16][PLD];  // staged P tile (f16, A-frag source)
  __shared__ float    Al[4][16];       // per-row alpha / 1/sum broadcast

  const int w  = threadIdx.x >> 5;
  const int l  = threadIdx.x & 31;
  const int b  = blockIdx.z;
  const int h  = blockIdx.y;
  const int q0 = blockIdx.x * 64 + w * 16;

  const int col  = l & 15;          // C-layout column / owned row index
  const int hf   = l >> 4;          // lane half
  const int rsel = hf << 3;         // C-layout row offset (0 or 8)
  const int row  = col;             // row owned for softmax stats
  const int cb   = hf << 4;         // owned column chunk (0 or 16)

  const _Float16* qbase = Q + (size_t)(b * TOK + q0) * INNER + h * DHEAD;
  v16h qa0 = load_fragA(qbase, INNER, 0);
  v16h qa1 = load_fragA(qbase, INNER, 32);

  const _Float16* kbase = Kh + (size_t)(b * TOK) * INNER + h * DHEAD;
  const _Float16* vbase = Vt + (size_t)(b * NHEADS + h) * DHEAD * TOK;
  const int* mrow = mask + b * TOK;

  v8f o[4];
#pragma unroll
  for (int j = 0; j < 4; ++j) o[j] = (v8f){0, 0, 0, 0, 0, 0, 0, 0};
  float mx = -1e30f;   // running max of owned row
  float sm = 0.0f;     // running sum of owned row

  const v8f zf = (v8f){0, 0, 0, 0, 0, 0, 0, 0};

  for (int kb = 0; kb < TOK; kb += 32) {
    // ---- scores: S[q, kb..kb+31] = Q (16x64) * K^T, fp32 accum
    const _Float16* kp0 = kbase + (size_t)kb * INNER;
    const _Float16* kp1 = kp0 + (size_t)16 * INNER;
    v8f s0 = wmma_f16(qa0, load_fragB(kp0, INNER, 0), zf);
    s0     = wmma_f16(qa1, load_fragB(kp0, INNER, 32), s0);
    v8f s1 = wmma_f16(qa0, load_fragB(kp1, INNER, 0), zf);
    s1     = wmma_f16(qa1, load_fragB(kp1, INNER, 32), s1);

    const int mv0 = mrow[kb + col];
    const int mv1 = mrow[kb + 16 + col];
    // scale + mask, stage S tile to LDS (C-layout -> row-major)
#pragma unroll
    for (int v = 0; v < 8; ++v) {
      float a = s0[v] * 0.125f;            // DIM_HEAD^-0.5
      float c = s1[v] * 0.125f;
      Sl[w][v + rsel][col]      = (mv0 == 0) ? 1e-9f : a;  // masked -> 1e-9
      Sl[w][v + rsel][col + 16] = (mv1 == 0) ? 1e-9f : c;
    }
    asm volatile("s_wait_dscnt 0" ::: "memory");

    // ---- row-owner softmax: lane handles 16 contiguous values of its row
    const float* srow = &Sl[w][row][cb];
    v4f x0 = *(const v4f*)(srow + 0);
    v4f x1 = *(const v4f*)(srow + 4);
    v4f x2 = *(const v4f*)(srow + 8);
    v4f x3 = *(const v4f*)(srow + 12);

    v4f mm;
#pragma unroll
    for (int i = 0; i < 4; ++i)
      mm[i] = fmaxf(fmaxf(x0[i], x1[i]), fmaxf(x2[i], x3[i]));
    float pm = fmaxf(fmaxf(mm[0], mm[1]), fmaxf(mm[2], mm[3]));
    float tm = fmaxf(pm, __shfl_xor(pm, 16, 32));   // combine halves
    float nm = fmaxf(mx, tm);
    float alpha = __expf(mx - nm);
    mx = nm;

    v8h ph0, ph1;
    float ps = 0.0f;
#pragma unroll
    for (int i = 0; i < 4; ++i) {
      float p0 = __expf(x0[i] - nm); ps += p0; ph0[i]     = (_Float16)p0;
      float p1 = __expf(x1[i] - nm); ps += p1; ph0[4 + i] = (_Float16)p1;
      float p2 = __expf(x2[i] - nm); ps += p2; ph1[i]     = (_Float16)p2;
      float p3 = __expf(x3[i] - nm); ps += p3; ph1[4 + i] = (_Float16)p3;
    }
    float rs = ps + __shfl_xor(ps, 16, 32);         // combine halves
    sm = sm * alpha + rs;

    // stage P (f16) and broadcast alpha (both halves write identical values)
    *(v8h*)&Pl[w][row][cb]     = ph0;
    *(v8h*)&Pl[w][row][cb + 8] = ph1;
    Al[w][row] = alpha;
    asm volatile("s_wait_dscnt 0" ::: "memory");

    // ---- rescale O accumulators with per-row alpha (contiguous b128 loads)
    v4f al0 = *(const v4f*)&Al[w][rsel];
    v4f al1 = *(const v4f*)&Al[w][rsel + 4];
#pragma unroll
    for (int j = 0; j < 4; ++j)
#pragma unroll
      for (int v = 0; v < 8; ++v)
        o[j][v] *= (v < 4) ? al0[v] : al1[v - 4];

    // ---- O += P (16x32) * V (32x64); V^T rows contiguous in key index
    v16h pf = load_fragA(&Pl[w][0][0], PLD, 0);
#pragma unroll
    for (int j = 0; j < 4; ++j) {
      v16h vf = load_fragB(vbase + (size_t)(16 * j) * TOK, TOK, kb);
      o[j] = wmma_f16(pf, vf, o[j]);
    }
  }

  // ---- normalize (broadcast 1/sum through LDS) and store f16 output
  Al[w][row] = 1.0f / sm;
  asm volatile("s_wait_dscnt 0" ::: "memory");
  v4f n0 = *(const v4f*)&Al[w][rsel];
  v4f n1 = *(const v4f*)&Al[w][rsel + 4];

  _Float16* aob = AO + (size_t)(b * TOK + q0) * INNER + h * DHEAD;
#pragma unroll
  for (int j = 0; j < 4; ++j)
#pragma unroll
    for (int v = 0; v < 8; ++v) {
      int r = v + rsel;
      float nf = (v < 4) ? n0[v] : n1[v - 4];
      aob[(size_t)r * INNER + 16 * j + col] = (_Float16)(o[j][v] * nf);
    }
}

// ---------------------------------------------------------------------------
extern "C" void kernel_launch(void* const* d_in, const int* in_sizes, int n_in,
                              void* d_out, int out_size, void* d_ws, size_t ws_size,
                              hipStream_t stream) {
  (void)in_sizes; (void)n_in; (void)out_size; (void)ws_size;
  const float* X    = (const float*)d_in[0];
  const int*   mask = (const int*)d_in[1];
  const float* Wq   = (const float*)d_in[2];
  const float* Wk   = (const float*)d_in[3];
  const float* Wv   = (const float*)d_in[4];
  const float* Wo   = (const float*)d_in[5];
  float* out = (float*)d_out;

  // workspace carve-up (48 MB total)
  char* ws = (char*)d_ws;
  _Float16* Xh  = (_Float16*)ws; ws += (size_t)MTOT * DMODEL * 2;
  _Float16* Wqh = (_Float16*)ws; ws += (size_t)INNER * DMODEL * 2;
  _Float16* Wkh = (_Float16*)ws; ws += (size_t)INNER * DMODEL * 2;
  _Float16* Wvh = (_Float16*)ws; ws += (size_t)INNER * DMODEL * 2;
  _Float16* Woh = (_Float16*)ws; ws += (size_t)DMODEL * INNER * 2;
  _Float16* Qh  = (_Float16*)ws; ws += (size_t)MTOT * INNER * 2;
  _Float16* Khp = (_Float16*)ws; ws += (size_t)MTOT * INNER * 2;
  _Float16* Vth = (_Float16*)ws; ws += (size_t)MTOT * INNER * 2;
  _Float16* AOh = (_Float16*)ws; ws += (size_t)MTOT * INNER * 2;

  // fp32 -> f16 conversions
  {
    int n4 = MTOT * DMODEL / 4;
    cvt_f32_to_f16_kernel<<<(n4 + 255) / 256, 256, 0, stream>>>(X, Xh, n4);
    int w4 = INNER * DMODEL / 4;
    cvt_f32_to_f16_kernel<<<(w4 + 255) / 256, 256, 0, stream>>>(Wq, Wqh, w4);
    cvt_f32_to_f16_kernel<<<(w4 + 255) / 256, 256, 0, stream>>>(Wk, Wkh, w4);
    cvt_f32_to_f16_kernel<<<(w4 + 255) / 256, 256, 0, stream>>>(Wv, Wvh, w4);
    cvt_f32_to_f16_kernel<<<(w4 + 255) / 256, 256, 0, stream>>>(Wo, Woh, w4);
  }

  dim3 gg(INNER / 256, MTOT / 64);   // (4, 64)
  gemm_wmma_kernel<0><<<gg, 256, 0, stream>>>(Xh, Wqh, (void*)Qh,  MTOT, INNER, DMODEL);
  gemm_wmma_kernel<0><<<gg, 256, 0, stream>>>(Xh, Wkh, (void*)Khp, MTOT, INNER, DMODEL);
  gemm_wmma_kernel<1><<<gg, 256, 0, stream>>>(Xh, Wvh, (void*)Vth, MTOT, INNER, DMODEL);

  attention_wmma_kernel<<<dim3(TOK / 64, NHEADS, BATCH), 128, 0, stream>>>(
      Qh, Khp, Vth, mask, AOh);

  gemm_wmma_kernel<2><<<gg, 256, 0, stream>>>(AOh, Woh, (void*)out, MTOT, DMODEL, INNER);
}
